// ScaledAttention_15015205667366
// MI455X (gfx1250) — matrix operand order repaired and checked by reading, not compiled, not run on hardware
//
#include <hip/hip_runtime.h>
#include <hip/hip_bf16.h>
#include <math.h>

// ---------------------------------------------------------------------------
// Causal attention (B=2, S=4096, DIM=ATTN_DIM=1024), fp32 in/out.
// Compute-bound (~189 GFLOP dense, <1 GB traffic) -> bf16 WMMA pipeline:
//   K1: QKV = X*W^T + b        (WMMA bf16, fp32 accum)  -> Q,K,V bf16 in ws
//   per batch (stream-serialized, S-buffer reused):
//   K2: S = Q*K^T              (WMMA, causal tile skip, async global->LDS)
//   K3: row softmax (scale 1/32 inside exp), P bf16 overlaid on S buffer
//   K4: O = P*V                (WMMA, causal K-loop, async P staging)
// Workspace: 3*16MB (QKV bf16) + 64MB (S fp32 / P bf16) = 112 MB.
// ---------------------------------------------------------------------------

typedef __bf16 bf16;
typedef __attribute__((ext_vector_type(16))) __bf16 v16bf;
typedef __attribute__((ext_vector_type(8)))  __bf16 v8bf;
typedef __attribute__((ext_vector_type(8)))  float  v8f;
typedef __attribute__((ext_vector_type(4)))  float  v4f;

#define DIMK  1024
#define SEQ   4096
#define NBATCH 2

#define BM 128
#define BN 128
#define BK 32
#define LDT 40   // LDS row stride (bf16) for 32-wide tiles: 80B, 16B-aligned, padded

static __device__ __forceinline__ bf16 tobf(float f) {
  union { float f; unsigned u; } a; a.f = f;
  unsigned u = a.u + 0x7FFFu + ((a.u >> 16) & 1u);     // round-to-nearest-even
  union { unsigned short s; bf16 b; } o; o.s = (unsigned short)(u >> 16);
  return o.b;
}

// Per ISA 10.2: flat addresses of LDS objects carry the allocation-relative
// LDS byte offset in their low 32 bits (aperture lives in addr[63:32]).
static __device__ __forceinline__ unsigned lds_off(const void* p) {
  return (unsigned)(unsigned long long)p;
}

// GLOBAL_LOAD_ASYNC_TO_LDS_B128, GVS mode: per-lane 16B  mem[saddr + voff] -> LDS[ldsa].
// Tracked by ASYNCcnt (ISA 10.7 / 15.18.3); waited with s_wait_asynccnt.
static __device__ __forceinline__ void async_ld_b128(unsigned ldsa, unsigned voff,
                                                     const void* sbase) {
  asm volatile("global_load_async_to_lds_b128 %0, %1, %2"
               :: "v"(ldsa), "v"(voff), "s"(sbase) : "memory");
}
static __device__ __forceinline__ void wait_async0() {
  asm volatile("s_wait_asynccnt 0" ::: "memory");
}

static __device__ __forceinline__ v16bf cat8(v8bf lo, v8bf hi) {
  v16bf r;
  #pragma unroll
  for (int i = 0; i < 8; ++i) { r[i] = lo[i]; r[8 + i] = hi[i]; }
  return r;
}

// A-matrix 16x32 bf16 fragment (ISA 7.12.2): lane L row M=L&15;
// lanes 0-15 hold K=0..7 then 16..23, lanes 16-31 hold K=8..15 then 24..31.
static __device__ __forceinline__ v16bf load_a_frag(const bf16* t, int lane) {
  const int m = lane & 15;
  const int h = (lane >> 4) << 3;                  // 0 or 8
  const bf16* p = t + m * LDT + h;
  return cat8(*(const v8bf*)p, *(const v8bf*)(p + 16));
}

// B-matrix 32x16 bf16 fragment, stored transposed in LDS as Bt[n][k]:
// lane L column N=L&15; lanes 0-15 hold K=0..7 then 16..23, lanes 16-31 hold K=8..15 then 24..31.
static __device__ __forceinline__ v16bf load_b_frag(const bf16* t, int lane) {
  const int n  = lane & 15;
  const int kb = (lane >> 4) << 3;                 // 0 or 8
  const bf16* p = t + n * LDT + kb;
  return cat8(*(const v8bf*)p, *(const v8bf*)(p + 16));
}

static __device__ __forceinline__ v8f wmma_bf16(v16bf a, v16bf b, v8f c) {
  return __builtin_amdgcn_wmma_f32_16x16x32_bf16(
      false, a, false, b, (short)0, c, false, false);
}

// --------------------------- K1: QKV projection ----------------------------
__global__ __launch_bounds__(256) void qkv_gemm_kernel(
    const float* __restrict__ x, const float* __restrict__ w,
    const float* __restrict__ bias, bf16* __restrict__ qo,
    bf16* __restrict__ ko, bf16* __restrict__ vo) {
  __shared__ bf16 At[BM * LDT];
  __shared__ bf16 Bt[BN * LDT];
  const int tid = threadIdx.x, lane = tid & 31, wid = tid >> 5;
  const int wm = wid & 1, wn = wid >> 1;           // 2x4 wave grid: 64x32 / wave
  const int m0 = blockIdx.x * BM;                  // rows over B*S = 8192
  const int e0 = blockIdx.y * BN;                  // cols over 3*1024 = 3072
  v8f acc[4][2];
  #pragma unroll
  for (int i = 0; i < 4; ++i) { v8f z = {}; acc[i][0] = z; acc[i][1] = z; }

  for (int k0 = 0; k0 < DIMK; k0 += BK) {
    __syncthreads();
    for (int i = tid; i < BM * 8; i += 256) {      // X tile: 128 x 32 fp32 -> bf16
      int r = i >> 3, c = (i & 7) << 2;
      v4f f = *(const v4f*)(x + (size_t)(m0 + r) * DIMK + k0 + c);
      bf16* d = &At[r * LDT + c];
      d[0] = tobf(f[0]); d[1] = tobf(f[1]); d[2] = tobf(f[2]); d[3] = tobf(f[3]);
    }
    for (int i = tid; i < BN * 8; i += 256) {      // W tile: 128 x 32 fp32 -> bf16
      int r = i >> 3, c = (i & 7) << 2;
      v4f f = *(const v4f*)(w + (size_t)(e0 + r) * DIMK + k0 + c);
      bf16* d = &Bt[r * LDT + c];
      d[0] = tobf(f[0]); d[1] = tobf(f[1]); d[2] = tobf(f[2]); d[3] = tobf(f[3]);
    }
    __syncthreads();
    v16bf b0 = load_b_frag(&Bt[(wn * 32 +  0) * LDT], lane);
    v16bf b1 = load_b_frag(&Bt[(wn * 32 + 16) * LDT], lane);
    #pragma unroll
    for (int mt = 0; mt < 4; ++mt) {
      v16bf a = load_a_frag(&At[(wm * 64 + mt * 16) * LDT], lane);
      acc[mt][0] = wmma_bf16(a, b0, acc[mt][0]);
      acc[mt][1] = wmma_bf16(a, b1, acc[mt][1]);
    }
  }

  const int seg = e0 / DIMK;                       // 0=Q 1=K 2=V (128 | 1024)
  bf16* dst = (seg == 0) ? qo : (seg == 1) ? ko : vo;
  const int c0 = e0 - seg * DIMK + wn * 32;
  #pragma unroll
  for (int nt = 0; nt < 2; ++nt) {
    const int ecol = c0 + nt * 16 + (lane & 15);
    const float bv = bias[seg * DIMK + ecol];
    #pragma unroll
    for (int mt = 0; mt < 4; ++mt)
      #pragma unroll
      for (int i = 0; i < 8; ++i) {
        int row = m0 + wm * 64 + mt * 16 + i + ((lane >> 4) << 3);
        dst[(size_t)row * DIMK + ecol] = tobf(acc[mt][nt][i] + bv);
      }
  }
}

// --------------------------- K2: S = Q * K^T -------------------------------
__global__ __launch_bounds__(256) void qk_gemm_kernel(
    const bf16* __restrict__ qb, const bf16* __restrict__ kb,
    float* __restrict__ sbuf) {
  if (blockIdx.y > blockIdx.x) return;             // causal: skip upper tiles
  __shared__ bf16 At[BM * LDT];
  __shared__ bf16 Bt[BN * LDT];
  const int tid = threadIdx.x, lane = tid & 31, wid = tid >> 5;
  const int wm = wid & 1, wn = wid >> 1;
  const int m0 = blockIdx.x * BM;                  // query rows
  const int n0 = blockIdx.y * BN;                  // key rows
  v8f acc[4][2];
  #pragma unroll
  for (int i = 0; i < 4; ++i) { v8f z = {}; acc[i][0] = z; acc[i][1] = z; }

  for (int k0 = 0; k0 < DIMK; k0 += BK) {
    __syncthreads();
    // Async global->LDS staging (no VGPR round trip); per-lane 16B chunks.
    for (int i = tid; i < BM * 4; i += 256) {
      int r = i >> 2, c = (i & 3) << 3;
      async_ld_b128(lds_off(&At[r * LDT + c]),
                    (unsigned)(((size_t)(m0 + r) * DIMK + k0 + c) * sizeof(bf16)), qb);
    }
    for (int i = tid; i < BN * 4; i += 256) {
      int r = i >> 2, c = (i & 3) << 3;
      async_ld_b128(lds_off(&Bt[r * LDT + c]),
                    (unsigned)(((size_t)(n0 + r) * DIMK + k0 + c) * sizeof(bf16)), kb);
    }
    if (k0 + BK < DIMK) {                          // global_prefetch_b8 next slab
      __builtin_prefetch(qb + (size_t)(m0 + (tid & 127)) * DIMK + k0 + BK, 0, 1);
      __builtin_prefetch(kb + (size_t)(n0 + (tid & 127)) * DIMK + k0 + BK, 0, 1);
    }
    wait_async0();
    __syncthreads();
    v16bf b0 = load_b_frag(&Bt[(wn * 32 +  0) * LDT], lane);
    v16bf b1 = load_b_frag(&Bt[(wn * 32 + 16) * LDT], lane);
    #pragma unroll
    for (int mt = 0; mt < 4; ++mt) {
      v16bf a = load_a_frag(&At[(wm * 64 + mt * 16) * LDT], lane);
      acc[mt][0] = wmma_bf16(a, b0, acc[mt][0]);
      acc[mt][1] = wmma_bf16(a, b1, acc[mt][1]);
    }
  }
  #pragma unroll
  for (int nt = 0; nt < 2; ++nt)
    #pragma unroll
    for (int mt = 0; mt < 4; ++mt)
      #pragma unroll
      for (int i = 0; i < 8; ++i) {
        int row = m0 + wm * 64 + mt * 16 + i + ((lane >> 4) << 3);
        int col = n0 + wn * 32 + nt * 16 + (lane & 15);
        sbuf[(size_t)row * SEQ + col] = acc[mt][nt][i];
      }
}

// --------------------- K3: masked row softmax -> P (bf16) ------------------
__global__ __launch_bounds__(256) void softmax_kernel(float* __restrict__ sbuf) {
  __shared__ float rowv[SEQ];
  __shared__ float red[8];
  const int q = blockIdx.x;
  const int tid = threadIdx.x, lane = tid & 31, wid = tid >> 5;
  float* sr = sbuf + (size_t)q * SEQ;

  float mx = -INFINITY;
  for (int j = tid; j < SEQ; j += 256) {
    float v = sr[j];
    rowv[j] = v;
    if (j <= q) mx = fmaxf(mx, v);
  }
  #pragma unroll
  for (int o = 16; o > 0; o >>= 1) mx = fmaxf(mx, __shfl_xor(mx, o, 32));
  if (lane == 0) red[wid] = mx;
  __syncthreads();
  if (wid == 0) {
    float t = (lane < 8) ? red[lane] : -INFINITY;
    #pragma unroll
    for (int o = 4; o > 0; o >>= 1) t = fmaxf(t, __shfl_xor(t, o, 32));
    if (lane == 0) red[0] = t;
  }
  __syncthreads();
  mx = red[0];
  __syncthreads();                                 // red[] reused below

  const float scale = 0.03125f;                    // 1/sqrt(1024)
  float sum = 0.f;
  for (int j = tid; j < SEQ; j += 256) {
    float e = (j <= q) ? __expf((rowv[j] - mx) * scale) : 0.f;
    rowv[j] = e;
    sum += e;
  }
  #pragma unroll
  for (int o = 16; o > 0; o >>= 1) sum += __shfl_xor(sum, o, 32);
  if (lane == 0) red[wid] = sum;
  __syncthreads();
  if (wid == 0) {
    float t = (lane < 8) ? red[lane] : 0.f;
    #pragma unroll
    for (int o = 4; o > 0; o >>= 1) t += __shfl_xor(t, o, 32);
    if (lane == 0) red[0] = t;
  }
  __syncthreads();
  const float inv = 1.0f / red[0];

  bf16* pr = (bf16*)sr;                            // P overlays S: stride 2*SEQ bf16
  for (int j = tid; j < SEQ; j += 256) pr[j] = tobf(rowv[j] * inv);
}

// --------------------------- K4: O = P * V ---------------------------------
__global__ __launch_bounds__(256) void pv_gemm_kernel(
    const bf16* __restrict__ pbuf, const bf16* __restrict__ vb,
    float* __restrict__ out) {
  __shared__ bf16 At[BM * LDT];
  __shared__ bf16 Bt[BN * LDT];
  const int tid = threadIdx.x, lane = tid & 31, wid = tid >> 5;
  const int wm = wid & 1, wn = wid >> 1;
  const int m0 = blockIdx.x * BM;                  // query rows
  const int n0 = blockIdx.y * BN;                  // output feature cols
  const int kend = (blockIdx.x + 1) * BM;          // causal: P rows zero past q
  v8f acc[4][2];
  #pragma unroll
  for (int i = 0; i < 4; ++i) { v8f z = {}; acc[i][0] = z; acc[i][1] = z; }

  for (int k0 = 0; k0 < kend; k0 += BK) {
    __syncthreads();
    for (int i = tid; i < BM * 4; i += 256) {      // P tile async (stride 2*SEQ)
      int r = i >> 2, c = (i & 3) << 3;
      async_ld_b128(lds_off(&At[r * LDT + c]),
                    (unsigned)(((size_t)(m0 + r) * (2 * SEQ) + k0 + c) * sizeof(bf16)),
                    pbuf);
    }
    for (int i = tid; i < BK * 16; i += 256) {     // V tile, transposed into LDS
      int j = i >> 4, c = (i & 15) << 3;
      v8bf t = *(const v8bf*)(vb + (size_t)(k0 + j) * DIMK + n0 + c);
      #pragma unroll
      for (int u = 0; u < 8; ++u) Bt[(c + u) * LDT + j] = t[u];
    }
    if (k0 + BK < kend)
      __builtin_prefetch(vb + (size_t)(k0 + BK + (tid & 31)) * DIMK + n0, 0, 1);
    wait_async0();
    __syncthreads();
    v16bf b0 = load_b_frag(&Bt[(wn * 32 +  0) * LDT], lane);
    v16bf b1 = load_b_frag(&Bt[(wn * 32 + 16) * LDT], lane);
    #pragma unroll
    for (int mt = 0; mt < 4; ++mt) {
      v16bf a = load_a_frag(&At[(wm * 64 + mt * 16) * LDT], lane);
      acc[mt][0] = wmma_bf16(a, b0, acc[mt][0]);
      acc[mt][1] = wmma_bf16(a, b1, acc[mt][1]);
    }
  }
  #pragma unroll
  for (int nt = 0; nt < 2; ++nt)
    #pragma unroll
    for (int mt = 0; mt < 4; ++mt)
      #pragma unroll
      for (int i = 0; i < 8; ++i) {
        int row = m0 + wm * 64 + mt * 16 + i + ((lane >> 4) << 3);
        int col = n0 + wn * 32 + nt * 16 + (lane & 15);
        out[(size_t)row * DIMK + col] = acc[mt][nt][i];
      }
}

// ---------------------------------------------------------------------------
extern "C" void kernel_launch(void* const* d_in, const int* in_sizes, int n_in,
                              void* d_out, int out_size, void* d_ws, size_t ws_size,
                              hipStream_t stream) {
  (void)in_sizes; (void)n_in; (void)out_size; (void)ws_size;
  const float* x  = (const float*)d_in[0];         // [2,4096,1024]
  const float* w  = (const float*)d_in[1];         // [3072,1024]
  const float* bq = (const float*)d_in[2];         // [3072]
  float* out = (float*)d_out;                      // [2,4096,1024]

  const size_t perMat = (size_t)NBATCH * SEQ * DIMK;     // 8M elems
  bf16* Q = (bf16*)d_ws;
  bf16* K = Q + perMat;
  bf16* V = K + perMat;
  float* Sb = (float*)((char*)d_ws + 3 * perMat * sizeof(bf16));  // 64 MB, reused

  dim3 blk(256);
  qkv_gemm_kernel<<<dim3((NBATCH * SEQ) / BM, (3 * DIMK) / BN), blk, 0, stream>>>(
      x, w, bq, Q, K, V);

  for (int b = 0; b < NBATCH; ++b) {
    const bf16* Qb = Q + (size_t)b * SEQ * DIMK;
    const bf16* Kb = K + (size_t)b * SEQ * DIMK;
    const bf16* Vb = V + (size_t)b * SEQ * DIMK;
    qk_gemm_kernel<<<dim3(SEQ / BM, SEQ / BN), blk, 0, stream>>>(Qb, Kb, Sb);
    softmax_kernel<<<SEQ, blk, 0, stream>>>(Sb);
    pv_gemm_kernel<<<dim3(SEQ / BM, DIMK / BN), blk, 0, stream>>>(
        (const bf16*)Sb, Vb, out + (size_t)b * SEQ * DIMK);
  }
}
